// LogisticRegressionRBF_26422638805219
// MI455X (gfx1250) — matrix operand order repaired and checked by reading, not compiled, not run on hardware
//
#include <hip/hip_runtime.h>

// ---------------------------------------------------------------------------
// RBF logistic regression forward, fused for MI455X (gfx1250, wave32, WMMA).
//   cross[k,n] = sum_m x[k,m]*c[n,m]      (f16 WMMA, f32 accumulate)
//   phi[k,n]   = exp(-(xsq[k]+csq[n]-2*cross))
//   out[k]     = sigmoid(sum_n phi[k,n]*W[n] + b)
// Compute-bound (AI ~1700 FLOP/byte):
//  - f16 WMMA / f32 accumulate for the 16384x4096x256 GEMM
//  - 32 rows per wave (2 A tiles in VGPRs): each B fragment feeds 2 WMMAs on
//    independent accumulator chains (ILP for the XDL pipe, half the DS reads)
//  - B tiles DMA'd into LDS by the Tensor Data Mover (tensor_load_to_lds,
//    TENSORcnt double-buffering) instead of per-thread loads
//  - N split 4-ways across blocks (grid=256) for machine fill
// ---------------------------------------------------------------------------

typedef __attribute__((ext_vector_type(16))) _Float16     v16h;
typedef __attribute__((ext_vector_type(8)))  _Float16     v8h;
typedef __attribute__((ext_vector_type(8)))  float        v8f;
typedef __attribute__((ext_vector_type(4)))  unsigned int v4u;
typedef __attribute__((ext_vector_type(8)))  int          v8i;
typedef __attribute__((ext_vector_type(4)))  int          v4i;

#define K_SAMPLES 16384
#define N_CENTERS 4096
#define M_FEAT    256
#define NSPLIT    4
#define ROWS_PER_BLOCK 256            // 8 waves x 32 rows
#define N_PER_SPLIT (N_CENTERS / NSPLIT)

// --- Stage 0: f32 -> f16 conversion + row squared-norms (one row per block) --
__global__ __launch_bounds__(256) void rbf_prep(const float* __restrict__ src,
                                                _Float16* __restrict__ dst,
                                                float* __restrict__ sq) {
  const int row = blockIdx.x;
  const int t   = threadIdx.x;              // 256 threads == M_FEAT
  const float v = src[(size_t)row * M_FEAT + t];
  dst[(size_t)row * M_FEAT + t] = (_Float16)v;

  float p = v * v;
  #pragma unroll
  for (int off = 16; off > 0; off >>= 1)    // wave32 reduction
    p += __shfl_xor(p, off, 32);

  __shared__ float red[8];
  if ((t & 31) == 0) red[t >> 5] = p;
  __syncthreads();
  if (t == 0) {
    float s = 0.f;
    #pragma unroll
    for (int i = 0; i < 8; ++i) s += red[i];
    sq[row] = s;
  }
}

// --- TDM: DMA one 16 x 256 f16 tile (8 KB) from global into LDS -------------
// D# per CDNA5 ISA ch.8: group0 {count=1, lds_addr, global_addr[56:0], type=2},
// group1 {data_size=2B, tensor_dim0=M, tensor_dim1=N, tile=256x16, stride=M}.
// This toolchain (clang-23 / therock-10.0 headers) exposes the 6-arg builtin:
// (uint32x4 g0, int32x8 g1, int32x4, int32x4, int32x8, i32 cpol).
__device__ __forceinline__ void tdm_load_tile(const _Float16* gsrc,
                                              unsigned lds_off) {
  const unsigned long long ga = (unsigned long long)(uintptr_t)gsrc;
  v4u g0;
  g0[0] = 1u;                                            // count=1 (valid D#)
  g0[1] = lds_off;                                       // lds_addr (bytes)
  g0[2] = (unsigned)(ga & 0xffffffffu);                  // global_addr[31:0]
  g0[3] = (unsigned)((ga >> 32) & 0x01ffffffu)           // global_addr[56:32]
        | (2u << 30);                                    // type = 2 ("image")
  v8i g1;
  g1[0] = (int)(1u << 16);                               // data_size=1 -> 2 B
  g1[1] = (int)(((unsigned)M_FEAT & 0xffffu) << 16);     // tensor_dim0[15:0]
  g1[2] = (int)((((unsigned)M_FEAT >> 16) & 0xffffu)     // tensor_dim0[31:16]
        | (((unsigned)N_CENTERS & 0xffffu) << 16));      // tensor_dim1[15:0]
  g1[3] = (int)((((unsigned)N_CENTERS >> 16) & 0xffffu)  // tensor_dim1[31:16]
        | (((unsigned)M_FEAT) << 16));                   // tile_dim0 = 256
  g1[4] = 16;                                            // tile_dim1=16, dim2=0
  g1[5] = M_FEAT;                                        // dim0_stride[31:0]
  g1[6] = 0;                                             // stride hi / dim1_stride
  g1[7] = 0;
  v4i z4 = {};
  v8i z8 = {};
  __builtin_amdgcn_tensor_load_to_lds(g0, g1, z4, z4, z8, 0); // 2-D: extras zero
}

// --- Stage 1: fused WMMA GEMM + RBF + weighted partial reduction ------------
__global__ __launch_bounds__(256) void rbf_main(
    const _Float16* __restrict__ xh, const _Float16* __restrict__ ch,
    const float* __restrict__ xsq, const float* __restrict__ csq,
    const float* __restrict__ W, float* __restrict__ partial) {
  __shared__ _Float16 ldsB[2][16 * M_FEAT];   // double-buffered 8 KB B tiles

  const int tid   = threadIdx.x;
  const int wave  = tid >> 5;
  const int lane  = tid & 31;
  const int l16   = lane & 15;              // column / row-in-tile selector
  const int half  = lane >> 4;              // half-wave (K-split per ISA layout)
  const int rowblk = blockIdx.x / NSPLIT;
  const int split  = blockIdx.x % NSPLIT;
  const int k0     = rowblk * ROWS_PER_BLOCK + wave * 32;
  const int nbase  = split * N_PER_SPLIT;

  // A fragments: 16-bit A 16x32, contiguous k order (not the ISA table order). Lane (l16, half) holds row M=l16,
// halves 0..15 = K = 16*half + 0..15 (matches B fragment order).
  v16h afrag[2][8];
  #pragma unroll
  for (int s = 0; s < 2; ++s) {
    const _Float16* arow = xh + (size_t)(k0 + 16 * s + l16) * M_FEAT;
    #pragma unroll
    for (int j = 0; j < 8; ++j) {
      v8h lo = *(const v8h*)(arow + j * 32 + 16 * half);
      v8h hi = *(const v8h*)(arow + j * 32 + 16 * half + 8);
      #pragma unroll
      for (int i = 0; i < 8; ++i) {
        afrag[s][j][i] = lo[i]; afrag[s][j][i + 8] = hi[i];
      }
    }
  }

  // C/D layout: VGPR v of lane (l16, half) = element (row = 8*half + v, col = l16)
  float xs0[8], xs1[8], acc0[8], acc1[8];
  #pragma unroll
  for (int v = 0; v < 8; ++v) {
    xs0[v] = xsq[k0 + 8 * half + v];
    xs1[v] = xsq[k0 + 16 + 8 * half + v];
    acc0[v] = 0.f; acc1[v] = 0.f;
  }

  const unsigned ldsb0 = (unsigned)(uintptr_t)&ldsB[0][0];
  const unsigned ldsb1 = (unsigned)(uintptr_t)&ldsB[1][0];

  constexpr int T = N_PER_SPLIT / 16;       // 64 tiles per split
  if (wave == 0)                            // prime the pipeline: tile 0
    tdm_load_tile(ch + (size_t)nbase * M_FEAT, ldsb0);

  for (int i = 0; i < T; ++i) {
    // All waves finished reading the buffer about to be overwritten.
    __syncthreads();
    if (wave == 0) {
      if (i + 1 < T) {                      // DMA next tile, wait current
        tdm_load_tile(ch + (size_t)(nbase + (i + 1) * 16) * M_FEAT,
                      ((i + 1) & 1) ? ldsb1 : ldsb0);
        __builtin_amdgcn_s_wait_tensorcnt(1);
      } else {
        __builtin_amdgcn_s_wait_tensorcnt(0);
      }
    }
    __syncthreads();                        // tile i published to all waves

    const _Float16* Bt = &ldsB[i & 1][0];
    const int n0 = nbase + i * 16;

    // 8 K-chunks; each B fragment (from LDS) feeds 2 WMMAs on independent
    // accumulator chains. B 32x16 layout: lane (l16, half) holds column n=l16,
    // halves = K = 16*half + 0..15 of each 32-chunk.
    v8f c0 = {}, c1 = {};
    #pragma unroll
    for (int j = 0; j < 8; ++j) {
      v16h bfrag = *(const v16h*)(&Bt[l16 * M_FEAT + j * 32 + 16 * half]);
      c0 = __builtin_amdgcn_wmma_f32_16x16x32_f16(
          false, afrag[0][j], false, bfrag, (short)0, c0, false, false);
      c1 = __builtin_amdgcn_wmma_f32_16x16x32_f16(
          false, afrag[1][j], false, bfrag, (short)0, c1, false, false);
    }

    // Fused RBF + weight: no phi materialization.
    const int   n  = n0 + l16;
    const float cs = csq[n];
    const float wn = W[n];
    #pragma unroll
    for (int v = 0; v < 8; ++v) {
      const float d0 = xs0[v] + cs - 2.0f * c0[v];
      const float d1 = xs1[v] + cs - 2.0f * c1[v];
      acc0[v] += wn * __expf(-d0);
      acc1[v] += wn * __expf(-d1);
    }
  }

  // Reduce over the 16 lanes (= 16 columns) of each half-wave.
  #pragma unroll
  for (int v = 0; v < 8; ++v) {
    #pragma unroll
    for (int off = 8; off > 0; off >>= 1) {
      acc0[v] += __shfl_xor(acc0[v], off, 16);
      acc1[v] += __shfl_xor(acc1[v], off, 16);
    }
  }

  if (l16 == 0) {
    float* p = partial + (size_t)split * K_SAMPLES;
    #pragma unroll
    for (int v = 0; v < 8; ++v) {
      p[k0 + 8 * half + v]      = acc0[v];
      p[k0 + 16 + 8 * half + v] = acc1[v];
    }
  }
}

// --- Stage 2: combine N-splits, add bias, sigmoid ---------------------------
__global__ __launch_bounds__(256) void rbf_finish(const float* __restrict__ partial,
                                                  const float* __restrict__ bptr,
                                                  float* __restrict__ out) {
  const int k = blockIdx.x * 256 + threadIdx.x;
  float s = bptr[0];
  #pragma unroll
  for (int i = 0; i < NSPLIT; ++i) s += partial[(size_t)i * K_SAMPLES + k];
  out[k] = 1.0f / (1.0f + __expf(-s));
}

// ---------------------------------------------------------------------------
extern "C" void kernel_launch(void* const* d_in, const int* in_sizes, int n_in,
                              void* d_out, int out_size, void* d_ws, size_t ws_size,
                              hipStream_t stream) {
  const float* x  = (const float*)d_in[0];   // [K, M]
  const float* xb = (const float*)d_in[1];   // [N, M]
  const float* W  = (const float*)d_in[2];   // [1, N]
  const float* b  = (const float*)d_in[3];   // [1]
  float* out = (float*)d_out;                // [K, 1]

  char* ws = (char*)d_ws;
  _Float16* xh  = (_Float16*)ws;                                         // 8 MB
  _Float16* chb = (_Float16*)(ws + (size_t)K_SAMPLES * M_FEAT * 2);      // 2 MB
  float*    xsq = (float*)(ws + (size_t)(K_SAMPLES + N_CENTERS) * M_FEAT * 2);
  float*    csq = xsq + K_SAMPLES;
  float*    partial = csq + N_CENTERS;       // NSPLIT * K floats (256 KB)

  rbf_prep<<<K_SAMPLES, 256, 0, stream>>>(x,  xh,  xsq);
  rbf_prep<<<N_CENTERS, 256, 0, stream>>>(xb, chb, csq);
  rbf_main<<<(K_SAMPLES / ROWS_PER_BLOCK) * NSPLIT, 256, 0, stream>>>(
      xh, chb, xsq, csq, W, partial);
  rbf_finish<<<K_SAMPLES / 256, 256, 0, stream>>>(partial, b, out);
}
